// BahdanauAttention_18854906429853
// MI455X (gfx1250) — matrix operand order repaired and checked by reading, not compiled, not run on hardware
//
#include <hip/hip_runtime.h>
#include <hip/hip_bf16.h>
#include <math.h>

// ---------------------------------------------------------------------------
// BahdanauAttention fused implementation for gfx1250 (MI455X, wave32, WMMA)
//
// Sizes: B=128, N=196, ENC=2048, DEC=512, ATT=512
// Main GEMM: [25088 x 2048] x [2048 x 512] via v_wmma_f32_16x16x32_bf16,
// fused with bias/att2 add, ReLU, dot(W_full) reduction -> e[25088].
// softmax(e) is invariant to b_full, so b_full is dropped.
//
// Main kernel: double-buffered LDS A-stage (bf16 fragment order), float4
// global prefetch + b64 LDS stores, 1 barrier per 64-wide K chunk.
// K loop kept at unroll(1) so WMMA accumulators stay register-resident
// in place (no cross-copy v_movs / hazard nops).
// ---------------------------------------------------------------------------

typedef __attribute__((ext_vector_type(16))) __bf16 v16bf;
typedef __attribute__((ext_vector_type(8)))  float  v8f;

#define BATCH 128
#define NPOS  196
#define ENC   2048
#define DEC   512
#define ATT   512
#define MROWS 25088            // BATCH * NPOS
#define ROWS_PER_WG 32
#define MAIN_THREADS 512       // 16 waves of 32
#define KITERS (ENC / 64)      // 32 chunks of 64 K

__device__ __forceinline__ unsigned short f32_to_bf16_bits(float f) {
    union { float f; unsigned u; } in; in.f = f;
    unsigned r = in.u + 0x7FFFu + ((in.u >> 16) & 1u);   // round-to-nearest-even
    return (unsigned short)(r >> 16);
}

__device__ __forceinline__ __bf16 f32_to_bf16(float f) {
    union { unsigned short s; __bf16 b; } out;
    out.s = f32_to_bf16_bits(f);
    return out.b;
}

__device__ __forceinline__ unsigned pack2_bf16(float a, float b) {
    return (unsigned)f32_to_bf16_bits(a) | ((unsigned)f32_to_bf16_bits(b) << 16);
}

// ---------------------------------------------------------------------------
// Kernel 0: att2[b][a] = decoder_hidden[b] . W_dec[:,a] + b_dec[a]   (128x512)
// ---------------------------------------------------------------------------
__global__ void __launch_bounds__(256) att2_kernel(
    const float* __restrict__ dh, const float* __restrict__ W_dec,
    const float* __restrict__ b_dec, float* __restrict__ att2)
{
    int idx = blockIdx.x * 256 + threadIdx.x;      // 0..65535
    int b = idx >> 9;
    int a = idx & 511;
    float s = b_dec[a];
    const float* dhb = dh + (size_t)b * DEC;
    #pragma unroll 8
    for (int k = 0; k < DEC; ++k)
        s += dhb[k] * W_dec[(size_t)k * ATT + a];
    att2[idx] = s;
}

// ---------------------------------------------------------------------------
// Kernel 1: repack W_enc (f32 [2048][512]) into bf16 WMMA B-fragment order.
// Wf flat index = ((ct*64 + s)*32 + lane)*16 + i
//   B 32x16 layout (contiguous k order, same as A stage): half = lane/16, n = lane%16,
//   element i -> K = s*32 + half*16 + i, col = ct*16 + n
// ---------------------------------------------------------------------------
__global__ void __launch_bounds__(256) pack_wenc_kernel(
    const float* __restrict__ W_enc, __bf16* __restrict__ Wf)
{
    int idx = blockIdx.x * 256 + threadIdx.x;      // 0..1048575
    int i    = idx & 15;
    int lane = (idx >> 4) & 31;
    int s    = (idx >> 9) & 63;
    int ct   = idx >> 15;
    int k    = s * 32 + (lane >> 4) * 16 + i;
    int col  = ct * 16 + (lane & 15);
    Wf[idx] = f32_to_bf16(W_enc[(size_t)k * ATT + col]);
}

// ---------------------------------------------------------------------------
// Kernel 2: fused GEMM + bias + relu + dot(W_full) -> e[row]
// Grid: 784 WGs x 512 threads. WG owns 32 rows; wave w owns cols [32w,32w+32).
// ---------------------------------------------------------------------------
__global__ void __launch_bounds__(MAIN_THREADS) fused_e_kernel(
    const float* __restrict__ enc,   const float* __restrict__ att2,
    const float* __restrict__ b_enc, const float* __restrict__ W_full,
    const __bf16* __restrict__ Wf,   float* __restrict__ e_out)
{
    // Double-buffered A stage in fragment order:
    //   flat d = ((rt*2 + sub)*32 + lane)*16 + i   (2048 per buffer)
    __shared__ __align__(32) __bf16 sA[2 * ROWS_PER_WG * 64];   // 8KB
    __shared__ float sE[ROWS_PER_WG];

    const int tid  = threadIdx.x;
    const int wave = tid >> 5;
    const int lane = tid & 31;
    const int hf   = lane >> 4;      // wave half (0: lanes 0-15, 1: 16-31)
    const int lm   = lane & 15;
    const int rowBase = blockIdx.x * ROWS_PER_WG;

    if (tid < ROWS_PER_WG) sE[tid] = 0.0f;

    // Per-thread A staging slot: 4 consecutive fragment elements (8B in LDS),
    // which map to 4 consecutive K values (16B) in global memory.
    const int d0  = tid * 4;
    const int i0  = d0 & 15;                    // 0,4,8,12
    const int fl  = (d0 >> 4) & 31;             // fragment lane
    const int sub = (d0 >> 9) & 1;
    const int rt  = d0 >> 10;
    const int m   = rt * 16 + (fl & 15);
    // A 16x32 layout: k = sub*32 + (fl/16)*16 + i
    const int kstart = sub * 32 + ((fl >> 4) << 4) + i0;
    const float* aptr = enc + (size_t)(rowBase + m) * ENC + kstart;

    // Per-wave B fragment base (32 columns = 2 col tiles)
    const __bf16* wf0 = Wf + (size_t)((wave * 2 + 0) * 64) * 512 + lane * 16;
    const __bf16* wf1 = Wf + (size_t)((wave * 2 + 1) * 64) * 512 + lane * 16;

    // Individually named accumulators -> in-place v_wmma accumulation
    v8f acc00 = (v8f){0.f,0.f,0.f,0.f,0.f,0.f,0.f,0.f};
    v8f acc01 = (v8f){0.f,0.f,0.f,0.f,0.f,0.f,0.f,0.f};
    v8f acc10 = (v8f){0.f,0.f,0.f,0.f,0.f,0.f,0.f,0.f};
    v8f acc11 = (v8f){0.f,0.f,0.f,0.f,0.f,0.f,0.f,0.f};

    float4 cur = *(const float4*)aptr;          // prefetch chunk 0
    int buf = 0;

    // steady state: unconditional prefetch of chunk ki+1; no unrolling so the
    // accumulators have one def-chain per body and stay in place.
    #pragma unroll 1
    for (int ki = 0; ki < KITERS - 1; ++ki) {
        uint2 pk;
        pk.x = pack2_bf16(cur.x, cur.y);
        pk.y = pack2_bf16(cur.z, cur.w);
        *(uint2*)&sA[buf + d0] = pk;

        float4 nxt = *(const float4*)(aptr + (size_t)(ki + 1) * 64);

        __syncthreads();   // sA[buf] stores visible; prior reads of sA[buf] done

        const __bf16* ab = &sA[buf];
        #pragma unroll
        for (int s2 = 0; s2 < 2; ++s2) {
            v16bf a0 = *(const v16bf*)&ab[((0 * 2 + s2) * 32 + lane) * 16];
            v16bf a1 = *(const v16bf*)&ab[((1 * 2 + s2) * 32 + lane) * 16];
            v16bf b0 = *(const v16bf*)&wf0[(size_t)(ki * 2 + s2) * 512];
            v16bf b1 = *(const v16bf*)&wf1[(size_t)(ki * 2 + s2) * 512];
            acc00 = __builtin_amdgcn_wmma_f32_16x16x32_bf16(false, a0, false, b0, (short)0, acc00, false, false);
            acc01 = __builtin_amdgcn_wmma_f32_16x16x32_bf16(false, a0, false, b1, (short)0, acc01, false, false);
            acc10 = __builtin_amdgcn_wmma_f32_16x16x32_bf16(false, a1, false, b0, (short)0, acc10, false, false);
            acc11 = __builtin_amdgcn_wmma_f32_16x16x32_bf16(false, a1, false, b1, (short)0, acc11, false, false);
        }

        cur = nxt;
        buf ^= 2048;
    }
    // peeled final chunk
    {
        uint2 pk;
        pk.x = pack2_bf16(cur.x, cur.y);
        pk.y = pack2_bf16(cur.z, cur.w);
        *(uint2*)&sA[buf + d0] = pk;
        __syncthreads();
        const __bf16* ab = &sA[buf];
        #pragma unroll
        for (int s2 = 0; s2 < 2; ++s2) {
            v16bf a0 = *(const v16bf*)&ab[((0 * 2 + s2) * 32 + lane) * 16];
            v16bf a1 = *(const v16bf*)&ab[((1 * 2 + s2) * 32 + lane) * 16];
            v16bf b0 = *(const v16bf*)&wf0[(size_t)((KITERS - 1) * 2 + s2) * 512];
            v16bf b1 = *(const v16bf*)&wf1[(size_t)((KITERS - 1) * 2 + s2) * 512];
            acc00 = __builtin_amdgcn_wmma_f32_16x16x32_bf16(false, a0, false, b0, (short)0, acc00, false, false);
            acc01 = __builtin_amdgcn_wmma_f32_16x16x32_bf16(false, a0, false, b1, (short)0, acc01, false, false);
            acc10 = __builtin_amdgcn_wmma_f32_16x16x32_bf16(false, a1, false, b0, (short)0, acc10, false, false);
            acc11 = __builtin_amdgcn_wmma_f32_16x16x32_bf16(false, a1, false, b1, (short)0, acc11, false, false);
        }
    }

    // Epilogue: combined = relu(att1 + b_enc + att2[b]); partial = combined*W_full
    // C/D layout: elem r -> row = rt*16 + hf*8 + r, col = 32*wave + ct*16 + lm
    #pragma unroll
    for (int rt2 = 0; rt2 < 2; ++rt2) {
        #pragma unroll
        for (int ct = 0; ct < 2; ++ct) {
            const int col = wave * 32 + ct * 16 + lm;
            const float wf = W_full[col];
            const float be = b_enc[col];
            #pragma unroll
            for (int r = 0; r < 8; ++r) {
                int row  = rt2 * 16 + hf * 8 + r;
                int grow = rowBase + row;
                int b    = grow / NPOS;
                float av = (rt2 == 0)
                    ? (ct == 0 ? acc00[r] : acc01[r])
                    : (ct == 0 ? acc10[r] : acc11[r]);
                float v  = av + be + att2[(size_t)b * ATT + col];
                v = v > 0.f ? v : 0.f;
                v *= wf;
                // sum over the 16 columns held by this half-wave
                v += __shfl_xor(v, 1, 32);
                v += __shfl_xor(v, 2, 32);
                v += __shfl_xor(v, 4, 32);
                v += __shfl_xor(v, 8, 32);
                if (lm == 0) atomicAdd(&sE[row], v);   // ds_add_f32
            }
        }
    }
    __syncthreads();
    if (tid < ROWS_PER_WG) e_out[rowBase + tid] = sE[tid];
}

// ---------------------------------------------------------------------------
// Kernel 3: softmax over N=196 per batch (b_full dropped: shift-invariant)
// ---------------------------------------------------------------------------
__global__ void __launch_bounds__(256) softmax_kernel(
    const float* __restrict__ e, float* __restrict__ alpha)
{
    __shared__ float sred[8];
    const int b = blockIdx.x;
    const int t = threadIdx.x;
    float v = (t < NPOS) ? e[(size_t)b * NPOS + t] : -INFINITY;

    float m = v;
    m = fmaxf(m, __shfl_xor(m, 16, 32));
    m = fmaxf(m, __shfl_xor(m,  8, 32));
    m = fmaxf(m, __shfl_xor(m,  4, 32));
    m = fmaxf(m, __shfl_xor(m,  2, 32));
    m = fmaxf(m, __shfl_xor(m,  1, 32));
    if ((t & 31) == 0) sred[t >> 5] = m;
    __syncthreads();
    m = sred[0];
    #pragma unroll
    for (int i = 1; i < 8; ++i) m = fmaxf(m, sred[i]);
    __syncthreads();

    float ex = (t < NPOS) ? __expf(v - m) : 0.f;
    float s = ex;
    s += __shfl_xor(s, 16, 32);
    s += __shfl_xor(s,  8, 32);
    s += __shfl_xor(s,  4, 32);
    s += __shfl_xor(s,  2, 32);
    s += __shfl_xor(s,  1, 32);
    if ((t & 31) == 0) sred[t >> 5] = s;
    __syncthreads();
    s = 0.f;
    #pragma unroll
    for (int i = 0; i < 8; ++i) s += sred[i];

    if (t < NPOS) alpha[(size_t)b * NPOS + t] = ex / s;
}

// ---------------------------------------------------------------------------
// Kernel 4: awe[b][c] = sum_n alpha[b][n] * enc[b][n][c]   (streaming pass)
// ---------------------------------------------------------------------------
__global__ void __launch_bounds__(256) wsum_kernel(
    const float* __restrict__ enc, const float* __restrict__ alpha,
    float* __restrict__ awe)
{
    __shared__ float sa[NPOS];
    const int b = blockIdx.y;
    const int c = blockIdx.x * 256 + threadIdx.x;
    if (threadIdx.x < NPOS) sa[threadIdx.x] = alpha[(size_t)b * NPOS + threadIdx.x];
    __syncthreads();
    const float* base = enc + (size_t)b * NPOS * ENC + c;
    float s = 0.f;
    #pragma unroll 4
    for (int n = 0; n < NPOS; ++n)
        s += sa[n] * base[(size_t)n * ENC];
    awe[(size_t)b * ENC + c] = s;
}

// ---------------------------------------------------------------------------
// Launch
// ---------------------------------------------------------------------------
extern "C" void kernel_launch(void* const* d_in, const int* in_sizes, int n_in,
                              void* d_out, int out_size, void* d_ws, size_t ws_size,
                              hipStream_t stream) {
    const float* enc    = (const float*)d_in[0];   // [128,196,2048]
    const float* dh     = (const float*)d_in[1];   // [128,512]
    const float* W_enc  = (const float*)d_in[2];   // [2048,512]
    const float* b_enc  = (const float*)d_in[3];   // [512]
    const float* W_dec  = (const float*)d_in[4];   // [512,512]
    const float* b_dec  = (const float*)d_in[5];   // [512]
    const float* W_full = (const float*)d_in[6];   // [512]
    // d_in[7] = b_full: softmax is shift-invariant, unused.

    float* awe   = (float*)d_out;                  // [128,2048]
    float* alpha = awe + (size_t)BATCH * ENC;      // [128,196]

    // workspace layout (all 32B-aligned offsets)
    float*  att2 = (float*)d_ws;                   // 65536 f32
    float*  e    = att2 + BATCH * ATT;             // 25088 f32
    __bf16* Wf   = (__bf16*)(e + MROWS);           // 1048576 bf16 (2MB)

    att2_kernel     <<<(BATCH * ATT) / 256, 256, 0, stream>>>(dh, W_dec, b_dec, att2);
    pack_wenc_kernel<<<(ENC * ATT) / 256,   256, 0, stream>>>(W_enc, Wf);
    fused_e_kernel  <<<MROWS / ROWS_PER_WG, MAIN_THREADS, 0, stream>>>(
                        enc, att2, b_enc, W_full, Wf, e);
    softmax_kernel  <<<BATCH, 256, 0, stream>>>(e, alpha);
    wsum_kernel     <<<dim3(ENC / 256, BATCH), 256, 0, stream>>>(enc, alpha, awe);
}
